// Transformer_82325933130224
// MI455X (gfx1250) — matrix operand order repaired and checked by reading, not compiled, not run on hardware
//
#include <hip/hip_runtime.h>

// Model constants (match reference)
#define Dm 512
#define Sm 1024
#define Bm 2
#define Hm 8
#define DHm 64
#define FFm 2048
#define Vm 32000
#define LEm 2
#define LDm 2
#define BSm (Bm * Sm)   // 2048 rows of activations

typedef __attribute__((ext_vector_type(2)))  float  v2f;
typedef __attribute__((ext_vector_type(8)))  float  v8f;
typedef __bf16 bf16;
typedef __attribute__((ext_vector_type(8)))  __bf16 v8bf;
typedef __attribute__((ext_vector_type(16))) __bf16 v16bf;

enum { GF_TRANSB = 1, GF_RELU = 2, GF_CAUSAL = 4 };

// ---------------------------------------------------------------------------
// fp32 WMMA batched GEMM (attention scores + A@V): exact fp32 path.
// One wave per 32x32 C tile, 4x v_wmma_f32_16x16x4_f32 per K-step of 4.
// ---------------------------------------------------------------------------
__global__ __launch_bounds__(32)
void gemm32x32_wmma_f32(const float* __restrict__ A, const float* __restrict__ B,
                        float* __restrict__ C, const float* __restrict__ bias,
                        const float* __restrict__ resid,
                        int K, int lda, int ldb, int ldc,
                        long long oAb, long long oAh,
                        long long oBb, long long oBh,
                        long long oCb, long long oCh,
                        int Hdiv, float alpha, int flags)
{
    const int zb = blockIdx.z / Hdiv;
    const int zh = blockIdx.z % Hdiv;
    A += zb * oAb + zh * oAh;
    B += zb * oBb + zh * oBh;
    C += zb * oCb + zh * oCh;
    if (resid) resid += zb * oCb + zh * oCh;

    const int row0 = blockIdx.y * 32;
    const int col0 = blockIdx.x * 32;
    const int lane = threadIdx.x & 31;
    const int half = lane >> 4;
    const int r    = lane & 15;

    v8f acc00 = {}, acc01 = {}, acc10 = {}, acc11 = {};

    const float* Ar0 = A + (long long)(row0 + r) * lda;
    const float* Ar1 = A + (long long)(row0 + 16 + r) * lda;

    if (flags & GF_TRANSB) {
        const float* Bc0 = B + (long long)(col0 + r) * ldb;
        const float* Bc1 = B + (long long)(col0 + 16 + r) * ldb;
        for (int k = 0; k < K; k += 4) {
            const int kb = k + 2 * half;
            v2f a0 = *(const v2f*)(Ar0 + kb);
            v2f a1 = *(const v2f*)(Ar1 + kb);
            v2f b0 = *(const v2f*)(Bc0 + kb);
            v2f b1 = *(const v2f*)(Bc1 + kb);
            acc00 = __builtin_amdgcn_wmma_f32_16x16x4_f32(false, a0, false, b0, (short)0, acc00, false, false);
            acc01 = __builtin_amdgcn_wmma_f32_16x16x4_f32(false, a0, false, b1, (short)0, acc01, false, false);
            acc10 = __builtin_amdgcn_wmma_f32_16x16x4_f32(false, a1, false, b0, (short)0, acc10, false, false);
            acc11 = __builtin_amdgcn_wmma_f32_16x16x4_f32(false, a1, false, b1, (short)0, acc11, false, false);
        }
    } else {
        for (int k = 0; k < K; k += 4) {
            const int kb = k + 2 * half;
            v2f a0 = *(const v2f*)(Ar0 + kb);
            v2f a1 = *(const v2f*)(Ar1 + kb);
            const float* Bp = B + (long long)kb * ldb;
            v2f b0, b1;
            b0.x = Bp[col0 + r];        b0.y = Bp[ldb + col0 + r];
            b1.x = Bp[col0 + 16 + r];   b1.y = Bp[ldb + col0 + 16 + r];
            acc00 = __builtin_amdgcn_wmma_f32_16x16x4_f32(false, a0, false, b0, (short)0, acc00, false, false);
            acc01 = __builtin_amdgcn_wmma_f32_16x16x4_f32(false, a0, false, b1, (short)0, acc01, false, false);
            acc10 = __builtin_amdgcn_wmma_f32_16x16x4_f32(false, a1, false, b0, (short)0, acc10, false, false);
            acc11 = __builtin_amdgcn_wmma_f32_16x16x4_f32(false, a1, false, b1, (short)0, acc11, false, false);
        }
    }

    auto store_tile = [&](v8f acc, int rb, int cb) {
        const int col = cb + r;
        #pragma unroll
        for (int i = 0; i < 8; ++i) {
            const int row = rb + i + 8 * half;
            float v = acc[i] * alpha;
            if (bias)  v += bias[col];
            if (resid) v += resid[(long long)row * ldc + col];
            if (flags & GF_RELU)   v = v > 0.0f ? v : 0.0f;
            if (flags & GF_CAUSAL) v = (col > row) ? -__builtin_inff() : v;
            C[(long long)row * ldc + col] = v;
        }
    };
    store_tile(acc00, row0,      col0);
    store_tile(acc01, row0,      col0 + 16);
    store_tile(acc10, row0 + 16, col0);
    store_tile(acc11, row0 + 16, col0 + 16);
}

// ---------------------------------------------------------------------------
// Split-bf16 GEMM: C = act(Ahi@Bhi + Ahi@Blo + Alo@Bhi [+bias][+resid])
// A (hi/lo): [M,K] bf16 row-major, contiguous (lda == K).
// B (hi/lo): [N,K] bf16 row-major (pre-transposed), contiguous (ldb == K).
// One wave per 32x32 C tile, 12x v_wmma_f32_16x16x32_bf16 per K-step of 32.
// Fragment loads are contiguous 16B chunks per the ISA 16-bit A/B layouts:
//   A lane: K = k0+8*half .. +7  and  k0+16+8*half .. +7
//   B lane: K = k0+8*half .. +7  and  k0+16+8*half .. +7
// ---------------------------------------------------------------------------
__global__ __launch_bounds__(32)
void gemm32x32_wmma_bf16x3(const bf16* __restrict__ Ahi, const bf16* __restrict__ Alo,
                           const bf16* __restrict__ Bhi, const bf16* __restrict__ Blo,
                           float* __restrict__ C, const float* __restrict__ bias,
                           const float* __restrict__ resid,
                           int K, int ldc, int flags)
{
    const int row0 = blockIdx.y * 32;
    const int col0 = blockIdx.x * 32;
    const int lane = threadIdx.x & 31;
    const int half = lane >> 4;
    const int r    = lane & 15;

    v8f acc00 = {}, acc01 = {}, acc10 = {}, acc11 = {};

    const size_t aoff0 = (size_t)(row0 + r) * K;
    const size_t aoff1 = (size_t)(row0 + 16 + r) * K;
    const size_t boff0 = (size_t)(col0 + r) * K;
    const size_t boff1 = (size_t)(col0 + 16 + r) * K;

    #define LD_AFRAG(base, off, ka) \
        __builtin_shufflevector(*(const v8bf*)((base) + (off) + (ka)), \
                                *(const v8bf*)((base) + (off) + (ka) + 16), \
                                0,1,2,3,4,5,6,7,8,9,10,11,12,13,14,15)
    #define LD_BFRAG(base, off, kb) \
        __builtin_shufflevector(*(const v8bf*)((base) + (off) + (kb)), \
                                *(const v8bf*)((base) + (off) + (kb) + 16), \
                                0,1,2,3,4,5,6,7,8,9,10,11,12,13,14,15)

    for (int k = 0; k < K; k += 32) {
        const int ka = k + 8 * half;
        const int kb = k + 8 * half;
        v16bf a0h = LD_AFRAG(Ahi, aoff0, ka);
        v16bf a0l = LD_AFRAG(Alo, aoff0, ka);
        v16bf a1h = LD_AFRAG(Ahi, aoff1, ka);
        v16bf a1l = LD_AFRAG(Alo, aoff1, ka);
        v16bf b0h = LD_BFRAG(Bhi, boff0, kb);
        v16bf b0l = LD_BFRAG(Blo, boff0, kb);
        v16bf b1h = LD_BFRAG(Bhi, boff1, kb);
        v16bf b1l = LD_BFRAG(Blo, boff1, kb);

        acc00 = __builtin_amdgcn_wmma_f32_16x16x32_bf16(false, a0h, false, b0h, (short)0, acc00, false, false);
        acc00 = __builtin_amdgcn_wmma_f32_16x16x32_bf16(false, a0h, false, b0l, (short)0, acc00, false, false);
        acc00 = __builtin_amdgcn_wmma_f32_16x16x32_bf16(false, a0l, false, b0h, (short)0, acc00, false, false);

        acc01 = __builtin_amdgcn_wmma_f32_16x16x32_bf16(false, a0h, false, b1h, (short)0, acc01, false, false);
        acc01 = __builtin_amdgcn_wmma_f32_16x16x32_bf16(false, a0h, false, b1l, (short)0, acc01, false, false);
        acc01 = __builtin_amdgcn_wmma_f32_16x16x32_bf16(false, a0l, false, b1h, (short)0, acc01, false, false);

        acc10 = __builtin_amdgcn_wmma_f32_16x16x32_bf16(false, a1h, false, b0h, (short)0, acc10, false, false);
        acc10 = __builtin_amdgcn_wmma_f32_16x16x32_bf16(false, a1h, false, b0l, (short)0, acc10, false, false);
        acc10 = __builtin_amdgcn_wmma_f32_16x16x32_bf16(false, a1l, false, b0h, (short)0, acc10, false, false);

        acc11 = __builtin_amdgcn_wmma_f32_16x16x32_bf16(false, a1h, false, b1h, (short)0, acc11, false, false);
        acc11 = __builtin_amdgcn_wmma_f32_16x16x32_bf16(false, a1h, false, b1l, (short)0, acc11, false, false);
        acc11 = __builtin_amdgcn_wmma_f32_16x16x32_bf16(false, a1l, false, b1h, (short)0, acc11, false, false);
    }
    #undef LD_AFRAG
    #undef LD_BFRAG

    auto store_tile = [&](v8f acc, int rb, int cb) {
        const int col = cb + r;
        #pragma unroll
        for (int i = 0; i < 8; ++i) {
            const int row = rb + i + 8 * half;
            float v = acc[i];
            if (bias)  v += bias[col];
            if (resid) v += resid[(long long)row * ldc + col];
            if (flags & GF_RELU) v = v > 0.0f ? v : 0.0f;
            C[(long long)row * ldc + col] = v;
        }
    };
    store_tile(acc00, row0,      col0);
    store_tile(acc01, row0,      col0 + 16);
    store_tile(acc10, row0 + 16, col0);
    store_tile(acc11, row0 + 16, col0 + 16);
}

// ---------------------------------------------------------------------------
// f32 -> (bf16 hi, bf16 lo) splitters
// ---------------------------------------------------------------------------
__device__ __forceinline__ void split2(float x, bf16& h, bf16& l)
{
    bf16 hh = (bf16)x;
    h = hh;
    l = (bf16)(x - (float)hh);
}

// Elementwise split (A operands, layout preserved)
__global__ __launch_bounds__(256)
void cvt_split(const float* __restrict__ in, bf16* __restrict__ hi,
               bf16* __restrict__ lo, long long n)
{
    const long long t = (long long)blockIdx.x * 256 + threadIdx.x;
    if (t >= n) return;
    bf16 h, l;
    split2(in[t], h, l);
    hi[t] = h; lo[t] = l;
}

// Split + transpose: in [K,N] f32 row-major -> out [N,K] bf16 (hi/lo)
__global__ __launch_bounds__(256)
void cvt_split_T(const float* __restrict__ in, bf16* __restrict__ hi,
                 bf16* __restrict__ lo, int K, int N)
{
    const long long t = (long long)blockIdx.x * 256 + threadIdx.x;
    if (t >= (long long)K * N) return;
    const int n = (int)(t / K);
    const int k = (int)(t % K);
    bf16 h, l;
    split2(in[(long long)k * N + n], h, l);
    hi[t] = h; lo[t] = l;
}

// Split + repack per-head weights [H,D,DH] -> Bt [N=H*DH, K=D] bf16 (hi/lo)
__global__ __launch_bounds__(256)
void cvt_split_qkv(const float* __restrict__ w, bf16* __restrict__ hi,
                   bf16* __restrict__ lo)
{
    const int t = blockIdx.x * 256 + threadIdx.x;   // over (H*DH) * D = 512*512
    const int d = t & (Dm - 1);
    const int n = t >> 9;
    const int h = n >> 6;
    const int e = n & 63;
    bf16 hh, ll;
    split2(w[((long long)h * Dm + d) * DHm + e], hh, ll);
    hi[t] = hh; lo[t] = ll;
}

// ---------------------------------------------------------------------------
// Embedding + sinusoidal positional encoding -> x and enc
// ---------------------------------------------------------------------------
__global__ __launch_bounds__(256)
void embed_posenc(const int* __restrict__ tok, const float* __restrict__ emb,
                  float* __restrict__ x, float* __restrict__ enc)
{
    const int bs = blockIdx.x;
    const int t  = tok[bs];
    const int s  = bs % Sm;
    const float kf = -9.210340371976184f / (float)Dm;   // -ln(10000)/D
    for (int d = threadIdx.x; d < Dm; d += 256) {
        const int j = d >> 1;
        const float den = expf((float)(2 * j) * kf);
        const float ang = (float)s * den;
        const float pe  = (d & 1) ? cosf(ang) : sinf(ang);
        const float v   = emb[(long long)t * Dm + d] + pe;
        x[(long long)bs * Dm + d]   = v;
        enc[(long long)bs * Dm + d] = v;
    }
}

// ---------------------------------------------------------------------------
// Row LayerNorm, D=512 fixed. One block (128 threads) per row; safe in-place.
// ---------------------------------------------------------------------------
__global__ __launch_bounds__(128)
void layernorm512(const float* __restrict__ X, float* __restrict__ Y,
                  const float* __restrict__ g, const float* __restrict__ b)
{
    __shared__ float red[128];
    const float* xr = X + (long long)blockIdx.x * Dm;
    float* yr       = Y + (long long)blockIdx.x * Dm;
    const int tid = threadIdx.x;

    float v0 = xr[tid], v1 = xr[tid + 128], v2 = xr[tid + 256], v3 = xr[tid + 384];
    red[tid] = v0 + v1 + v2 + v3;
    __syncthreads();
    for (int s = 64; s > 0; s >>= 1) { if (tid < s) red[tid] += red[tid + s]; __syncthreads(); }
    const float mean = red[0] * (1.0f / Dm);
    __syncthreads();

    const float d0 = v0 - mean, d1 = v1 - mean, d2 = v2 - mean, d3 = v3 - mean;
    red[tid] = d0 * d0 + d1 * d1 + d2 * d2 + d3 * d3;
    __syncthreads();
    for (int s = 64; s > 0; s >>= 1) { if (tid < s) red[tid] += red[tid + s]; __syncthreads(); }
    const float rs = rsqrtf(red[0] * (1.0f / Dm) + 1e-5f);

    yr[tid]       = d0 * rs * g[tid]       + b[tid];
    yr[tid + 128] = d1 * rs * g[tid + 128] + b[tid + 128];
    yr[tid + 256] = d2 * rs * g[tid + 256] + b[tid + 256];
    yr[tid + 384] = d3 * rs * g[tid + 384] + b[tid + 384];
}

// ---------------------------------------------------------------------------
// Row softmax over ncols (-inf masked entries -> exp()=0)
// ---------------------------------------------------------------------------
__global__ __launch_bounds__(256)
void softmax_rows(float* __restrict__ S, int ncols)
{
    __shared__ float red[256];
    float* row = S + (long long)blockIdx.x * ncols;
    const int tid = threadIdx.x;

    float m = -__builtin_inff();
    for (int c = tid; c < ncols; c += 256) m = fmaxf(m, row[c]);
    red[tid] = m; __syncthreads();
    for (int s = 128; s > 0; s >>= 1) { if (tid < s) red[tid] = fmaxf(red[tid], red[tid + s]); __syncthreads(); }
    m = red[0];
    __syncthreads();

    float sum = 0.0f;
    for (int c = tid; c < ncols; c += 256) { float e = expf(row[c] - m); row[c] = e; sum += e; }
    red[tid] = sum; __syncthreads();
    for (int s = 128; s > 0; s >>= 1) { if (tid < s) red[tid] += red[tid + s]; __syncthreads(); }
    const float inv = 1.0f / red[0];
    for (int c = tid; c < ncols; c += 256) row[c] *= inv;
}

// ---------------------------------------------------------------------------
// Host orchestration
// ---------------------------------------------------------------------------
extern "C" void kernel_launch(void* const* d_in, const int* in_sizes, int n_in,
                              void* d_out, int out_size, void* d_ws, size_t ws_size,
                              hipStream_t stream)
{
    (void)in_sizes; (void)n_in; (void)out_size; (void)ws_size;

    const int*   tokens    = (const int*)  d_in[0];
    const float* emb       = (const float*)d_in[1];
    const float* enc_wq    = (const float*)d_in[2];
    const float* enc_wk    = (const float*)d_in[3];
    const float* enc_wv    = (const float*)d_in[4];
    const float* enc_uo_w  = (const float*)d_in[5];
    const float* enc_uo_b  = (const float*)d_in[6];
    const float* enc_ln1_g = (const float*)d_in[7];
    const float* enc_ln1_b = (const float*)d_in[8];
    const float* enc_ln2_g = (const float*)d_in[9];
    const float* enc_ln2_b = (const float*)d_in[10];
    const float* enc_ff_w1 = (const float*)d_in[11];
    const float* enc_ff_b1 = (const float*)d_in[12];
    const float* enc_ff_w2 = (const float*)d_in[13];
    const float* enc_ff_b2 = (const float*)d_in[14];
    const float* dec_sa_wq = (const float*)d_in[15];
    const float* dec_sa_wk = (const float*)d_in[16];
    const float* dec_sa_wv = (const float*)d_in[17];
    const float* dec_sa_uo_w = (const float*)d_in[18];
    const float* dec_sa_uo_b = (const float*)d_in[19];
    const float* dec_ca_wq = (const float*)d_in[20];
    const float* dec_ca_wk = (const float*)d_in[21];
    const float* dec_ca_wv = (const float*)d_in[22];
    const float* dec_ca_uo_w = (const float*)d_in[23];
    const float* dec_ca_uo_b = (const float*)d_in[24];
    const float* dec_ln1_g = (const float*)d_in[25];
    const float* dec_ln1_b = (const float*)d_in[26];
    const float* dec_ln2_g = (const float*)d_in[27];
    const float* dec_ln2_b = (const float*)d_in[28];
    const float* dec_ln3_g = (const float*)d_in[29];
    const float* dec_ln3_b = (const float*)d_in[30];
    const float* dec_ff_w1 = (const float*)d_in[31];
    const float* dec_ff_b1 = (const float*)d_in[32];
    const float* dec_ff_w2 = (const float*)d_in[33];
    const float* dec_ff_b2 = (const float*)d_in[34];
    const float* ln_enc_g  = (const float*)d_in[35];
    const float* ln_enc_b  = (const float*)d_in[36];
    const float* ln_dec_g  = (const float*)d_in[37];
    const float* ln_dec_b  = (const float*)d_in[38];
    const float* head_w    = (const float*)d_in[39];
    const float* head_b    = (const float*)d_in[40];

    // ---- Workspace layout ----
    float* ws = (float*)d_ws;
    size_t o = 0;
    float* x   = ws + o; o += (size_t)BSm * Dm;
    float* enc = ws + o; o += (size_t)BSm * Dm;
    float* h   = ws + o; o += (size_t)BSm * Dm;
    float* q   = ws + o; o += (size_t)BSm * Dm;
    float* kk  = ws + o; o += (size_t)BSm * Dm;
    float* vv  = ws + o; o += (size_t)BSm * Dm;
    float* ob  = ws + o; o += (size_t)BSm * Dm;
    float* ffb = ws + o; o += (size_t)BSm * FFm;
    float* sc  = ws + o; o += (size_t)Bm * Hm * Sm * Sm;
    // bf16 region (16B-aligned: o*4 is a multiple of 16 here)
    bf16* bws = (bf16*)(ws + o);
    size_t ob2 = 0;
    bf16* aHi = bws + ob2; ob2 += (size_t)BSm * FFm;      // max A: [2048,2048]
    bf16* aLo = bws + ob2; ob2 += (size_t)BSm * FFm;
    bf16* bHi = bws + ob2; ob2 += (size_t)Vm * Dm;        // max Bt: [32000,512]
    bf16* bLo = bws + ob2; ob2 += (size_t)Vm * Dm;

    auto GEMMF = [&](const float* A, const float* B, float* C,
                     const float* bias, const float* resid,
                     int M, int N, int K, int lda, int ldb, int ldc,
                     int Z, int Hdiv,
                     long long oAb, long long oAh, long long oBb, long long oBh,
                     long long oCb, long long oCh, float alpha, int flags) {
        dim3 grid(N / 32, M / 32, Z), block(32);
        gemm32x32_wmma_f32<<<grid, block, 0, stream>>>(A, B, C, bias, resid,
                                                       K, lda, ldb, ldc,
                                                       oAb, oAh, oBb, oBh, oCb, oCh,
                                                       Hdiv, alpha, flags);
    };
    // Split-bf16 GEMM: A pre-split in aHi/aLo [M,K]; Bt pre-split in bHi/bLo [N,K]
    auto GEMMB = [&](float* C, const float* bias, const float* resid,
                     int M, int N, int K, int ldc, int flags) {
        dim3 grid(N / 32, M / 32, 1), block(32);
        gemm32x32_wmma_bf16x3<<<grid, block, 0, stream>>>(aHi, aLo, bHi, bLo, C, bias,
                                                          resid, K, ldc, flags);
    };
    auto CVTA = [&](const float* src, long long n) {
        cvt_split<<<(unsigned)((n + 255) / 256), 256, 0, stream>>>(src, aHi, aLo, n);
    };
    auto CVTBT = [&](const float* src, int K, int N) {   // src [K,N] -> Bt [N,K]
        long long n = (long long)K * N;
        cvt_split_T<<<(unsigned)((n + 255) / 256), 256, 0, stream>>>(src, bHi, bLo, K, N);
    };
    auto CVTQKV = [&](const float* w) {                  // [H,D,DH] -> Bt [512,512]
        cvt_split_qkv<<<(Dm * Dm) / 256, 256, 0, stream>>>(w, bHi, bLo);
    };
    auto LN = [&](const float* src, float* dst, const float* g, const float* b) {
        layernorm512<<<BSm, 128, 0, stream>>>(src, dst, g, b);
    };

    // Full MHA block: qsrc/kvsrc are [BS,D] f32; resid_io is the residual stream.
    auto ATTN = [&](const float* qsrc, const float* kvsrc,
                    const float* wq, const float* wk, const float* wv,
                    const float* uoW, const float* uoB, float* resid_io) {
        // Projections via split-bf16 (q/k/v written f32 at [bs, h*64+e])
        if (qsrc == kvsrc) {
            CVTA(qsrc, (long long)BSm * Dm);
            CVTQKV(wq); GEMMB(q,  nullptr, nullptr, BSm, Dm, Dm, Dm, 0);
            CVTQKV(wk); GEMMB(kk, nullptr, nullptr, BSm, Dm, Dm, Dm, 0);
            CVTQKV(wv); GEMMB(vv, nullptr, nullptr, BSm, Dm, Dm, Dm, 0);
        } else {
            CVTA(qsrc, (long long)BSm * Dm);
            CVTQKV(wq); GEMMB(q,  nullptr, nullptr, BSm, Dm, Dm, Dm, 0);
            CVTA(kvsrc, (long long)BSm * Dm);
            CVTQKV(wk); GEMMB(kk, nullptr, nullptr, BSm, Dm, Dm, Dm, 0);
            CVTQKV(wv); GEMMB(vv, nullptr, nullptr, BSm, Dm, Dm, Dm, 0);
        }
        // Scores (exact fp32 WMMA): per (b,h) [S,DH]@[S,DH]^T, *sqrt(DH)=8, causal -inf
        GEMMF(q, kk, sc, nullptr, nullptr, Sm, Sm, DHm, Dm, Dm, Sm,
              Bm * Hm, Hm,
              (long long)Sm * Dm, DHm,
              (long long)Sm * Dm, DHm,
              (long long)Hm * Sm * Sm, (long long)Sm * Sm,
              8.0f, GF_TRANSB | GF_CAUSAL);
        softmax_rows<<<Bm * Hm * Sm, 256, 0, stream>>>(sc, Sm);
        // O = P@V (fp32 WMMA), written at [bs, h*64+e] (head concat free)
        GEMMF(sc, vv, ob, nullptr, nullptr, Sm, DHm, Sm, Sm, Dm, Dm,
              Bm * Hm, Hm,
              (long long)Hm * Sm * Sm, (long long)Sm * Sm,
              (long long)Sm * Dm, DHm,
              (long long)Sm * Dm, DHm,
              1.0f, 0);
        // Output projection + bias + residual (split-bf16)
        CVTA(ob, (long long)BSm * Dm);
        CVTBT(uoW, Dm, Dm);
        GEMMB(resid_io, uoB, resid_io, BSm, Dm, Dm, Dm, 0);
    };

    embed_posenc<<<BSm, 256, 0, stream>>>(tokens, emb, x, enc);

    // ---------------- Encoder ----------------
    for (int l = 0; l < LEm; ++l) {
        const size_t wo = (size_t)l * Hm * Dm * DHm;
        LN(enc, h, enc_ln1_g + l * Dm, enc_ln1_b + l * Dm);
        ATTN(h, h, enc_wq + wo, enc_wk + wo, enc_wv + wo,
             enc_uo_w + (size_t)l * Dm * Dm, enc_uo_b + l * Dm, enc);
        LN(enc, h, enc_ln2_g + l * Dm, enc_ln2_b + l * Dm);
        CVTA(h, (long long)BSm * Dm);
        CVTBT(enc_ff_w1 + (size_t)l * Dm * FFm, Dm, FFm);
        GEMMB(ffb, enc_ff_b1 + l * FFm, nullptr, BSm, FFm, Dm, FFm, GF_RELU);
        CVTA(ffb, (long long)BSm * FFm);
        CVTBT(enc_ff_w2 + (size_t)l * FFm * Dm, FFm, Dm);
        GEMMB(enc, enc_ff_b2 + l * Dm, enc, BSm, Dm, FFm, Dm, 0);
    }
    LN(enc, enc, ln_enc_g, ln_enc_b);

    // ---------------- Decoder ----------------
    for (int l = 0; l < LDm; ++l) {
        const size_t wo = (size_t)l * Hm * Dm * DHm;
        LN(x, h, dec_ln1_g + l * Dm, dec_ln1_b + l * Dm);
        ATTN(h, h, dec_sa_wq + wo, dec_sa_wk + wo, dec_sa_wv + wo,
             dec_sa_uo_w + (size_t)l * Dm * Dm, dec_sa_uo_b + l * Dm, x);
        LN(x, h, dec_ln2_g + l * Dm, dec_ln2_b + l * Dm);
        ATTN(h, enc, dec_ca_wq + wo, dec_ca_wk + wo, dec_ca_wv + wo,
             dec_ca_uo_w + (size_t)l * Dm * Dm, dec_ca_uo_b + l * Dm, x);
        LN(x, h, dec_ln3_g + l * Dm, dec_ln3_b + l * Dm);
        CVTA(h, (long long)BSm * Dm);
        CVTBT(dec_ff_w1 + (size_t)l * Dm * FFm, Dm, FFm);
        GEMMB(ffb, dec_ff_b1 + l * FFm, nullptr, BSm, FFm, Dm, FFm, GF_RELU);
        CVTA(ffb, (long long)BSm * FFm);
        CVTBT(dec_ff_w2 + (size_t)l * FFm * Dm, FFm, Dm);
        GEMMB(x, dec_ff_b2 + l * Dm, x, BSm, Dm, FFm, Dm, 0);
    }
    LN(x, x, ln_dec_g, ln_dec_b);

    // ---------------- LM head (split-bf16) ----------------
    CVTA(x, (long long)BSm * Dm);
    CVTBT(head_w, Dm, Vm);
    GEMMB((float*)d_out, head_b, nullptr, BSm, Vm, Dm, Vm, 0);
}
